// Decoder_76948634075330
// MI455X (gfx1250) — compile-verified, matrix operand order checked by reading, not run on hardware
//
#include <hip/hip_runtime.h>
#include <hip/hip_bf16.h>
#include <stdint.h>

typedef __attribute__((ext_vector_type(16))) _Float16 v16h;
typedef __attribute__((ext_vector_type(8)))  _Float16 v8h;
typedef __attribute__((ext_vector_type(8)))  float    v8f;

#define MPTS  1024
#define NPTS  1024
#define KOUT  64
#define FEAT  128
#define TOPK  4096
#define AFLAT 3072

// ---------------------------------------------------------------------------
// ws layout (bytes):
//   [0,1024)            uint hist[256]
//   [1024,1040)         ctrl: [256]=prefix/T, [257]=kremain, [258]=n_gt, [259]=n_eq
//   [4096, +2MB)        Nmat   f16 [M][N]   (N-branch, row-major)
//   [.. , +2MB)         Mcm    f16 [M][N]   (M-branch stored column-major: Mcm[m][n]=Mmat[n][m])
//   then uint2 gtbuf[4096], uint2 eqbuf[4096]
// ---------------------------------------------------------------------------

__device__ __forceinline__ unsigned long long spread3(unsigned x) {
    // spread 16 bits so bit p lands at bit 3p (classic 21-bit Morton spread)
    unsigned long long v = (unsigned long long)(x & 0xFFFFu);
    v = (v | (v << 32)) & 0x001F00000000FFFFull;
    v = (v | (v << 16)) & 0x001F0000FF0000FFull;
    v = (v | (v << 8))  & 0x100F00F00F00F00Full;
    v = (v | (v << 4))  & 0x10C30C30C30C30C3ull;
    v = (v | (v << 2))  & 0x1249249249249249ull;
    return v;
}

// ---------------------------------------------------------------------------
// Stage 1: one-hot heads as LDS row-gathers. block = one m-row, wave = 128 n's,
// lane l owns k = {2l, 2l+1}. Per plane: one conflict-free ds_load_b64 per table.
// ---------------------------------------------------------------------------
__global__ __launch_bounds__(256) void stage1_kernel(
    const int* __restrict__ a, const int* __restrict__ b,
    const float* __restrict__ Wn, const float* __restrict__ bn,
    const float* __restrict__ Wm, const float* __restrict__ bm,
    _Float16* __restrict__ Nh, _Float16* __restrict__ Mh)
{
    __shared__ float ldsWn[FEAT * KOUT];
    __shared__ float ldsWm[FEAT * KOUT];
    const int tid = threadIdx.x;
    for (int i = tid; i < FEAT * KOUT / 4; i += 256) {
        ((float4*)ldsWn)[i] = ((const float4*)Wn)[i];
        ((float4*)ldsWm)[i] = ((const float4*)Wm)[i];
    }
    __syncthreads();

    const int m    = blockIdx.x;
    const int wave = tid >> 5;
    const int lane = tid & 31;

    const int bx = b[3 * m + 0], by = b[3 * m + 1], bz = b[3 * m + 2];
    const float2 biasN = *(const float2*)(bn + 2 * lane);
    const float2 biasM = *(const float2*)(bm + 2 * lane);

    const int n0 = wave * 128;
    for (int n = n0; n < n0 + 128; ++n) {
        const unsigned fx = (unsigned)(a[3 * n + 0] + bx);
        const unsigned fy = (unsigned)(a[3 * n + 1] + by);
        const unsigned fz = (unsigned)(a[3 * n + 2] + bz);
        const unsigned long long mort =
            spread3(fx) | (spread3(fy) << 1) | (spread3(fz) << 2);

        float2 an = make_float2(0.f, 0.f);
        float2 am = make_float2(0.f, 0.f);
        #pragma unroll
        for (int p = 0; p < 16; ++p) {
            const unsigned tok = (unsigned)(mort >> (3 * p)) & 7u;
            const unsigned row = ((unsigned)p << 3) | tok;   // 8p + token
            const float2 wn = *(const float2*)(ldsWn + row * KOUT + 2 * lane);
            const float2 wm = *(const float2*)(ldsWm + row * KOUT + 2 * lane);
            an.x += wn.x; an.y += wn.y;
            am.x += wm.x; am.y += wm.y;
        }
        float sn = fmaxf(an.x + biasN.x, 0.f) + fmaxf(an.y + biasN.y, 0.f);
        float sm = fmaxf(am.x + biasM.x, 0.f) + fmaxf(am.y + biasM.y, 0.f);
        #pragma unroll
        for (int off = 16; off >= 1; off >>= 1) {
            sn += __shfl_xor(sn, off, 32);
            sm += __shfl_xor(sm, off, 32);
        }
        if (lane == 0) {
            Nh[m * NPTS + n] = (_Float16)(sn * (1.f / KOUT));
            Mh[m * NPTS + n] = (_Float16)(sm * (1.f / KOUT));  // Mcm[m][n] = Mmat[n][m]
        }
    }
}

// ---------------------------------------------------------------------------
// Stage 2: P = Nmat(f16) @ Mmat(f16) via v_wmma_f32_16x16x32_f16.
// Wave tile = 32x64 (2x4 16x16 tiles); block = 8 waves -> 128x128; grid 8x8.
// Both operands stream straight from L2 (2 MB each), two b128 loads per frag.
// ---------------------------------------------------------------------------
__global__ __launch_bounds__(256) void gemm_kernel(
    const _Float16* __restrict__ A,   // Nmat  [row][k]
    const _Float16* __restrict__ Bc,  // Mcm   [col][k]  (column-major B)
    float* __restrict__ P)
{
    const int tid  = threadIdx.x;
    const int lane = tid & 31;
    const int wave = tid >> 5;
    const int wr   = wave & 3;           // 4 row-groups
    const int wc   = wave >> 2;          // 2 col-groups
    const int baseRow = blockIdx.y * 128 + wr * 32;
    const int baseCol = blockIdx.x * 128 + wc * 64;
    const int r16   = lane & 15;
    const int half8 = (lane >> 4) << 3;  // 0 for lanes 0-15, 8 for lanes 16-31

    const v8f zacc = {0.f, 0.f, 0.f, 0.f, 0.f, 0.f, 0.f, 0.f};
    v8f acc[2][4];
    #pragma unroll
    for (int i = 0; i < 2; ++i)
        #pragma unroll
        for (int j = 0; j < 4; ++j) acc[i][j] = zacc;

    for (int kc = 0; kc < NPTS; kc += 32) {
        if (kc + 32 < NPTS) {  // global_prefetch_b8 for next k-chunk
            __builtin_prefetch(A  + (baseRow + r16) * NPTS + kc + 32, 0, 1);
            __builtin_prefetch(Bc + (baseCol + r16) * NPTS + kc + 32, 0, 1);
        }
        v16h af[2], bf[4];
        #pragma unroll
        for (int rt = 0; rt < 2; ++rt) {
            const _Float16* p = A + (baseRow + rt * 16 + r16) * NPTS + kc + half8;
            const v8h lo = *(const v8h*)p;
            const v8h hi = *(const v8h*)(p + 16);
            #pragma unroll
            for (int i = 0; i < 8; ++i) { af[rt][i] = lo[i]; af[rt][8 + i] = hi[i]; }
        }
        #pragma unroll
        for (int ct = 0; ct < 4; ++ct) {
            const _Float16* p = Bc + (baseCol + ct * 16 + r16) * NPTS + kc + half8;
            const v8h lo = *(const v8h*)p;
            const v8h hi = *(const v8h*)(p + 16);
            #pragma unroll
            for (int i = 0; i < 8; ++i) { bf[ct][i] = lo[i]; bf[ct][8 + i] = hi[i]; }
        }
        #pragma unroll
        for (int rt = 0; rt < 2; ++rt)
            #pragma unroll
            for (int ct = 0; ct < 4; ++ct)
                acc[rt][ct] = __builtin_amdgcn_wmma_f32_16x16x32_f16(
                    false, af[rt], false, bf[ct], (short)0, acc[rt][ct], false, false);
    }

    // C/D layout: VGPR j -> M = j + 8*(lane>=16), N = lane&15
    #pragma unroll
    for (int rt = 0; rt < 2; ++rt)
        #pragma unroll
        for (int ct = 0; ct < 4; ++ct)
            #pragma unroll
            for (int j = 0; j < 8; ++j)
                P[(baseRow + rt * 16 + half8 + j) * MPTS + baseCol + ct * 16 + r16] =
                    acc[rt][ct][j];
}

// ---------------------------------------------------------------------------
// Exact top-4096: radix select on order-preserving uint map, then bitonic sort.
// ---------------------------------------------------------------------------
__device__ __forceinline__ unsigned fmap(float f) {
    unsigned u = __float_as_uint(f);
    return (u & 0x80000000u) ? ~u : (u | 0x80000000u);
}

__global__ void select_init(unsigned* ws) {
    int t = threadIdx.x;                    // 256 threads
    ws[t] = 0;                              // hist
    if (t == 0) { ws[256] = 0; ws[257] = TOPK; ws[258] = 0; ws[259] = 0; }
}

__global__ __launch_bounds__(256) void select_hist(
    const float* __restrict__ P, unsigned* __restrict__ ws, int round)
{
    __shared__ unsigned lh[256];
    const int t = threadIdx.x;
    lh[t] = 0;
    __syncthreads();
    const int i = blockIdx.x * 256 + t;
    const unsigned u      = fmap(P[i]);
    const int      shift  = (3 - round) * 8;
    const unsigned prefix = ws[256];
    const unsigned mask   = round ? (0xFFFFFFFFu << (shift + 8)) : 0u;
    if ((u & mask) == prefix) atomicAdd(&lh[(u >> shift) & 255u], 1u);
    __syncthreads();
    if (lh[t]) atomicAdd(&ws[t], lh[t]);
}

__global__ void select_scan(unsigned* ws, int round) {
    const int shift = (3 - round) * 8;
    unsigned k = ws[257], cum = 0;
    for (int b = 255; b >= 0; --b) {
        const unsigned h = ws[b];
        if (cum + h >= k) {
            ws[257] = k - cum;
            ws[256] |= ((unsigned)b) << shift;
            break;
        }
        cum += h;
    }
    for (int i = 0; i < 256; ++i) ws[i] = 0;  // reset hist for next round
}

__global__ __launch_bounds__(256) void select_compact(
    const float* __restrict__ P, unsigned* __restrict__ ws,
    uint2* __restrict__ gtbuf, uint2* __restrict__ eqbuf)
{
    const int i = blockIdx.x * 256 + threadIdx.x;
    const unsigned T = ws[256];
    const unsigned u = fmap(P[i]);
    if (u > T) {
        unsigned g = atomicAdd(&ws[258], 1u);
        if (g < TOPK) gtbuf[g] = make_uint2(u, (unsigned)i);
    } else if (u == T) {
        unsigned e = atomicAdd(&ws[259], 1u);
        if (e < TOPK) eqbuf[e] = make_uint2(u, (unsigned)i);
    }
}

__global__ __launch_bounds__(1024) void topk_final(
    const unsigned* __restrict__ ws,
    const uint2* __restrict__ gtbuf, const uint2* __restrict__ eqbuf,
    const int* __restrict__ a_flat, int* __restrict__ out_g)
{
    __shared__ uint2 arr[TOPK];
    const int t = threadIdx.x;
    unsigned ngt = ws[258]; if (ngt > TOPK) ngt = TOPK;
    unsigned neq = ws[259]; if (neq > TOPK) neq = TOPK;

    for (int i = t; i < TOPK; i += 1024) {
        uint2 v;
        if ((unsigned)i < ngt)                 v = gtbuf[i];
        else if ((unsigned)i - ngt < neq)      v = eqbuf[(unsigned)i - ngt];
        else                                   v = make_uint2(0u, 0u);
        arr[i] = v;
    }
    __syncthreads();

    for (int k = 2; k <= TOPK; k <<= 1) {
        for (int j = k >> 1; j > 0; j >>= 1) {
            for (int i = t; i < TOPK; i += 1024) {
                const int p = i ^ j;
                if (p > i) {
                    const uint2 x = arr[i], y = arr[p];
                    const bool up = ((i & k) == 0);
                    const bool ybefore = (y.x > x.x) || (y.x == x.x && y.y < x.y);
                    if (ybefore == up) { arr[i] = y; arr[p] = x; }
                }
            }
            __syncthreads();
        }
    }

    for (int i = t; i < TOPK; i += 1024) {
        const uint2 v = arr[i];
        int g = 0;
        if (v.x > 0x80000000u)                 // val > 0.0f
            g = a_flat[v.y % (unsigned)AFLAT];
        out_g[i] = g;
    }
}

// ---------------------------------------------------------------------------
extern "C" void kernel_launch(void* const* d_in, const int* in_sizes, int n_in,
                              void* d_out, int out_size, void* d_ws, size_t ws_size,
                              hipStream_t stream)
{
    (void)in_sizes; (void)n_in; (void)out_size; (void)ws_size;
    const int*   a  = (const int*)d_in[0];
    const int*   b  = (const int*)d_in[1];
    const float* Wn = (const float*)d_in[2];
    const float* bn = (const float*)d_in[3];
    const float* Wm = (const float*)d_in[4];
    const float* bm = (const float*)d_in[5];

    float*    P    = (float*)d_out;                        // 1M f32
    char*     ws   = (char*)d_ws;
    unsigned* ctrl = (unsigned*)ws;                        // hist[256] + ctrl
    _Float16* Nh   = (_Float16*)(ws + 4096);
    _Float16* Mh   = Nh + (size_t)MPTS * NPTS;
    uint2*    gtb  = (uint2*)(ws + 4096 + (size_t)4 * 1024 * 1024);
    uint2*    eqb  = gtb + TOPK;

    stage1_kernel<<<MPTS, 256, 0, stream>>>(a, b, Wn, bn, Wm, bm, Nh, Mh);
    gemm_kernel<<<dim3(8, 8), 256, 0, stream>>>(Nh, Mh, P);

    select_init<<<1, 256, 0, stream>>>(ctrl);
    for (int r = 0; r < 4; ++r) {
        select_hist<<<(MPTS * MPTS) / 256, 256, 0, stream>>>(P, ctrl, r);
        select_scan<<<1, 1, 0, stream>>>(ctrl, r);
    }
    select_compact<<<(MPTS * MPTS) / 256, 256, 0, stream>>>(P, ctrl, gtb, eqb);
    topk_final<<<1, 1024, 0, stream>>>(ctrl, gtb, eqb, a,
                                       (int*)(P + (size_t)MPTS * MPTS));
}
